// LeftDecoder_9002251453268
// MI455X (gfx1250) — compile-verified, matrix operand order checked by reading, not run on hardware
//
#include <hip/hip_runtime.h>
#include <hip/hip_bf16.h>
#include <cstdint>
#include <cstddef>

// Problem dims (fixed by the reference)
constexpr int S  = 64;
constexpr int B  = 128;
constexpr int T  = 64;
constexpr int H  = 1024;
constexpr int E2 = 2048;
constexpr int AD = 1024;   // attention dim A
constexpr int WD = 512;    // embedding dim W
constexpr int OD = 1024;   // output dim O

typedef __bf16 bf16;
typedef __attribute__((ext_vector_type(16))) __bf16 v16bf;
typedef __attribute__((ext_vector_type(8)))  float  v8f;
typedef unsigned int v4u __attribute__((ext_vector_type(4)));
typedef int          v8i __attribute__((ext_vector_type(8)));
typedef int          v4i __attribute__((ext_vector_type(4)));

union FragBF { v16bf v; uint32_t u[8]; };

__device__ __forceinline__ float waveReduceSum(float v) {
#pragma unroll
  for (int m = 16; m > 0; m >>= 1) v += __shfl_xor(v, m, 32);
  return v;
}

// ---------------------------------------------------------------------------
// bf16 WMMA GEMM:  C[M,N] = act( A[M,K] * Wt[N,K]^T + bias[N] + add )
//
// block = 256 threads = 8 waves; block tile = 16(M) x 256(N).
// Wave tile = 16 x 32 (two 16x16 accumulators sharing one A fragment).
// The block's 16-row A stripe [16 x K] bf16 is DMA'd into LDS ONCE by the
// Tensor Data Mover (tensor_load_to_lds, D# per ISA section 8), so the K-loop
// runs with zero barriers: per 32-K chunk it is 2x ds_load_b128 (A frag),
// 4x global_load_b128 (two B frags), 2x v_wmma_f32_16x16x32_bf16.
// act: 0=none, 1=sigmoid, 2=tanh
// ---------------------------------------------------------------------------
__global__ void __launch_bounds__(256)
gemm_bf16_wmma(const bf16* __restrict__ Amat, const bf16* __restrict__ Wt,
               const float* __restrict__ bias, const float* add,
               float* C, bf16* Cbf, int M, int N, int K, int act)
{
  extern __shared__ char smemraw[];            // [16][K] bf16 ; LDS offset 0
  uint32_t* Alds = (uint32_t*)smemraw;         // 16 * (K/2) dwords

  const int tid  = threadIdx.x;
  const int lane = tid & 31;
  const int wv   = tid >> 5;                   // 0..7
  const int hl   = lane >> 4;                  // half-wave (0/1)
  const int ln   = lane & 15;
  const int m0   = blockIdx.y << 4;
  const int n0   = (blockIdx.x << 8) + (wv << 5);

  // ---- TDM: DMA the A stripe [16 x K] bf16 into LDS (wave 0 issues) ----
  if (wv == 0) {
    const uint64_t ga = (uint64_t)(uintptr_t)(Amat + (size_t)m0 * (size_t)K);
    const uint32_t k16 = (uint32_t)K & 0xFFFFu;
    // D# group 0: count=1 | lds_addr=0 | global_addr | type=2
    v4u g0 = { 1u,
               0u,
               (uint32_t)(ga & 0xFFFFFFFFull),
               (uint32_t)((ga >> 32) & 0x1FFFFFFull) | (2u << 30) };
    // D# group 1: data_size=2B; tensor_dim0=K; tensor_dim1=16;
    //             tile_dim0=K; tile_dim1=16; tensor_dim0_stride=K
    v8i g1 = { (int)(1u << 16),               // d0: data_size=1 (2 bytes)
               (int)(k16 << 16),              // d1: tensor_dim0[15:0] << 16
               (int)(16u << 16),              // d2: dim0 hi=0 | tensor_dim1=16
               (int)(k16 << 16),              // d3: dim1 hi=0 | tile_dim0=K
               16,                            // d4: tile_dim1=16, tile_dim2=0
               K,                             // d5: tensor_dim0_stride lo
               0, 0 };                        // d6,d7
    v4i g2 = { 0, 0, 0, 0 };
    v4i g3 = { 0, 0, 0, 0 };
#if defined(__clang_major__) && (__clang_major__ >= 23)
    v8i g4 = { 0, 0, 0, 0, 0, 0, 0, 0 };
    __builtin_amdgcn_tensor_load_to_lds(g0, g1, g2, g3, g4, 0);
#else
    __builtin_amdgcn_tensor_load_to_lds(g0, g1, g2, g3, 0);
#endif
    __builtin_amdgcn_s_wait_tensorcnt(0);
  }
  __syncthreads();

  const uint32_t* Wg = (const uint32_t*)Wt;
  const int nA = n0 + ln;                      // subtile 0 column
  const int nB = n0 + 16 + ln;                 // subtile 1 column
  const uint32_t rowbase = (uint32_t)ln * (uint32_t)(K >> 1);

  v8f acc0 = {};
  v8f acc1 = {};
  for (int kb = 0; kb < K; kb += 32) {
    if (kb + 64 < K) {   // stream-prefetch B two chunks ahead
      __builtin_prefetch(Wt + (size_t)nA * K + kb + 64, 0, 1);
    }
    FragBF a, b0, b1;
    const uint32_t abase  = rowbase + (uint32_t)(kb >> 1);
    const uint32_t wbase0 = ((uint32_t)nA * (uint32_t)K + (uint32_t)kb) >> 1;
    const uint32_t wbase1 = ((uint32_t)nB * (uint32_t)K + (uint32_t)kb) >> 1;
#pragma unroll
    for (int i = 0; i < 8; ++i) {
      // ISA 7.12.2 bf16 fragment: K offset for vgpr i, half-wave hl
      const int kd = (((i >> 2) << 4) + (hl << 3) + ((i & 3) << 1)) >> 1;
      a.u[i]  = Alds[abase + (uint32_t)kd];
      b0.u[i] = Wg[wbase0 + (uint32_t)kd];
      b1.u[i] = Wg[wbase1 + (uint32_t)kd];
    }
    acc0 = __builtin_amdgcn_wmma_f32_16x16x32_bf16(
        false, a.v, false, b0.v, (short)0, acc0, false, false);
    acc1 = __builtin_amdgcn_wmma_f32_16x16x32_bf16(
        false, a.v, false, b1.v, (short)0, acc1, false, false);
  }

  const float bnA = bias ? bias[nA] : 0.0f;
  const float bnB = bias ? bias[nB] : 0.0f;
#pragma unroll
  for (int j = 0; j < 8; ++j) {
    const int m = m0 + (hl << 3) + j;
    const size_t iA = (size_t)m * (size_t)N + (size_t)nA;
    const size_t iB = (size_t)m * (size_t)N + (size_t)nB;
    float vA = acc0[j] + bnA + (add ? add[iA] : 0.0f);
    float vB = acc1[j] + bnB + (add ? add[iB] : 0.0f);
    if (act == 1) {
      vA = 1.0f / (1.0f + __expf(-vA));
      vB = 1.0f / (1.0f + __expf(-vB));
    } else if (act == 2) {
      vA = tanhf(vA);
      vB = tanhf(vB);
    }
    C[iA] = vA;
    C[iB] = vB;
    if (Cbf) { Cbf[iA] = (bf16)vA; Cbf[iB] = (bf16)vB; }
  }
}

// ---------------------------------------------------------------------------
// Elementwise / attention kernels
// ---------------------------------------------------------------------------
__global__ void cvt_f32_bf16(const float* __restrict__ src, bf16* __restrict__ dst,
                             int n) {
  int i = blockIdx.x * 256 + threadIdx.x;
  if (i < n) dst[i] = (bf16)src[i];
}

__global__ void pack2_f32(const float* __restrict__ a, const float* __restrict__ b,
                          float* __restrict__ dst, int n) {
  int i = blockIdx.x * 256 + threadIdx.x;
  if (i < n) { dst[i] = a[i]; dst[n + i] = b[i]; }
}

__global__ void init_state(const float* __restrict__ s0, float* __restrict__ sf,
                           bf16* __restrict__ sb) {
  int i = blockIdx.x * 256 + threadIdx.x;   // B*H threads
  float v = s0[i];
  sf[i] = v;
  sb[i] = (bf16)v;
}

__global__ void embed_gather(const int* __restrict__ ys, const float* __restrict__ emb,
                             bf16* __restrict__ ye) {
  size_t i = (size_t)blockIdx.x * 256 + threadIdx.x;  // T*B*W threads
  const int r = (int)(i >> 9);          // / W (512)
  const int w = (int)(i & 511);
  ye[i] = (bf16)emb[(size_t)ys[r] * WD + w];
}

// e_raw[s,b] = exp( sum_a tanh(datt[b,a] + uh[s,b,a]) * a1w[a] + a1b )
__global__ void __launch_bounds__(256)
attn_energy(const float* __restrict__ datt, const float* __restrict__ uh,
            const float* __restrict__ a1w, const float* __restrict__ a1b,
            float* __restrict__ eraw) {
  const int lane = threadIdx.x & 31;
  const int gid  = (blockIdx.x << 3) + (threadIdx.x >> 5);  // s*B + b
  const int b    = gid & (B - 1);
  const float* du = uh   + (size_t)gid * AD;
  const float* dd = datt + (size_t)b   * AD;
  float acc = 0.f;
  for (int a = lane; a < AD; a += 32)
    acc += tanhf(dd[a] + du[a]) * a1w[a];
  acc = waveReduceSum(acc);
  if (lane == 0) eraw[gid] = __expf(acc + a1b[0]);
}

// inv[b] = 1 / sum_s eraw[s,b]   (one wave per b; S = 64 = 2*32)
__global__ void attn_colsum(const float* __restrict__ eraw, float* __restrict__ inv) {
  const int lane = threadIdx.x & 31;
  const int b    = (blockIdx.x << 3) + (threadIdx.x >> 5);
  float acc = eraw[lane * B + b] + eraw[(lane + 32) * B + b];
  acc = waveReduceSum(acc);
  if (lane == 0) inv[b] = 1.0f / acc;
}

// attend[b,e] = inv[b] * sum_s eraw[s,b] * xs_h[s,b,e]
__global__ void attn_ctx(const float* __restrict__ eraw, const float* __restrict__ inv,
                         const float* __restrict__ xsh, float* __restrict__ attf,
                         bf16* __restrict__ attb) {
  const int i = blockIdx.x * 256 + threadIdx.x;   // b*E2 + e
  const int b = i >> 11;
  const int e = i & (E2 - 1);
  float acc = 0.f;
#pragma unroll 4
  for (int s = 0; s < S; ++s)
    acc += eraw[s * B + b] * xsh[(((size_t)s * B + b) << 11) + e];
  acc *= inv[b];
  attf[i] = acc;
  attb[i] = (bf16)acc;
}

// rs = bf16( r * s ),  r = zr[b, H + h]
__global__ void rs_mul(const float* __restrict__ zr, const float* __restrict__ sf,
                       bf16* __restrict__ rsb) {
  const int i = blockIdx.x * 256 + threadIdx.x;   // B*H threads
  const int b = i >> 10;
  const int h = i & (H - 1);
  rsb[i] = (bf16)(zr[(size_t)b * (2 * H) + H + h] * sf[i]);
}

// s_new = (1-z)*s + z*h ; write fp32 state, bf16 state, and states output
__global__ void gru_update(const float* __restrict__ zr, const float* __restrict__ hc,
                           float* sf, bf16* __restrict__ sb,
                           float* __restrict__ states_out) {
  const int i = blockIdx.x * 256 + threadIdx.x;   // B*H threads
  const int b = i >> 10;
  const int h = i & (H - 1);
  const float z  = zr[(size_t)b * (2 * H) + h];
  const float sn = (1.0f - z) * sf[i] + z * hc[i];
  sf[i] = sn;
  sb[i] = (bf16)sn;
  states_out[i] = sn;
}

// maxout over pairs + ys_mask
__global__ void maxout_mask(const float* __restrict__ logit,
                            const float* __restrict__ ymask_t,
                            float* __restrict__ out_t) {
  const int i = blockIdx.x * 256 + threadIdx.x;   // B*O threads
  const int b = i >> 10;
  const int o = i & (OD - 1);
  const float v = fmaxf(logit[(size_t)b * (2 * OD) + 2 * o],
                        logit[(size_t)b * (2 * OD) + 2 * o + 1]);
  out_t[i] = v * ymask_t[b];
}

// ---------------------------------------------------------------------------
// Host side
// ---------------------------------------------------------------------------
extern "C" void kernel_launch(void* const* d_in, const int* in_sizes, int n_in,
                              void* d_out, int out_size, void* d_ws, size_t ws_size,
                              hipStream_t stream) {
  (void)in_sizes; (void)n_in; (void)out_size; (void)ws_size;

  // Inputs in setup_inputs() order
  const float* s_tm1   = (const float*)d_in[0];
  const float* xs_h    = (const float*)d_in[1];
  const int*   ys      = (const int*)  d_in[2];
  const float* uh      = (const float*)d_in[3];
  /* xs_mask d_in[4] unused (reference passes it into an unused slot) */
  const float* ys_mask = (const float*)d_in[5];
  const float* emb     = (const float*)d_in[6];
  const float* sa_w  = (const float*)d_in[7];   const float* sa_b = (const float*)d_in[8];
  const float* a1_w  = (const float*)d_in[9];   const float* a1_b = (const float*)d_in[10];
  const float* xz_w  = (const float*)d_in[11];  const float* xz_b = (const float*)d_in[12];
  const float* hz_w  = (const float*)d_in[13];  const float* hz_b = (const float*)d_in[14];
  const float* xr_w  = (const float*)d_in[15];  const float* xr_b = (const float*)d_in[16];
  const float* hr_w  = (const float*)d_in[17];  const float* hr_b = (const float*)d_in[18];
  const float* xh_w  = (const float*)d_in[19];  const float* xh_b = (const float*)d_in[20];
  const float* hh_w  = (const float*)d_in[21];  const float* hh_b = (const float*)d_in[22];
  const float* cz_w  = (const float*)d_in[23];  const float* cz_b = (const float*)d_in[24];
  const float* cr_w  = (const float*)d_in[25];  const float* cr_b = (const float*)d_in[26];
  const float* ch_w  = (const float*)d_in[27];  const float* ch_b = (const float*)d_in[28];
  const float* ls_w  = (const float*)d_in[29];  const float* ls_b = (const float*)d_in[30];
  const float* ly_w  = (const float*)d_in[31];  const float* ly_b = (const float*)d_in[32];
  const float* lc_w  = (const float*)d_in[33];  const float* lc_b = (const float*)d_in[34];

  float* out      = (float*)d_out;
  float* d_logits = out;                                // [T,B,O]
  float* d_states = out + (size_t)T * B * OD;           // [T,B,H]

  // Bump allocator over workspace
  char* wsb = (char*)d_ws;
  size_t off = 0;
  auto alloc = [&](size_t bytes) -> void* {
    void* p = wsb + off;
    off = (off + bytes + 255) & ~(size_t)255;
    return p;
  };
  auto abf = [&](size_t elems) -> bf16*  { return (bf16*) alloc(elems * 2); };
  auto af  = [&](size_t elems) -> float* { return (float*)alloc(elems * 4); };

  // bf16 weights (zr = [z-rows ; r-rows] packed)
  bf16* sa_bf  = abf((size_t)AD * H);
  bf16* zrx_bf = abf((size_t)2 * H * WD);
  bf16* zrh_bf = abf((size_t)2 * H * H);
  bf16* zrc_bf = abf((size_t)2 * H * E2);
  bf16* xh_bf  = abf((size_t)H * WD);
  bf16* hh_bf  = abf((size_t)H * H);
  bf16* ch_bf  = abf((size_t)H * E2);
  bf16* ls_bf  = abf((size_t)2 * OD * H);
  bf16* ly_bf  = abf((size_t)2 * OD * WD);
  bf16* lc_bf  = abf((size_t)2 * OD * E2);
  bf16* ye_bf  = abf((size_t)T * B * WD);
  // packed biases
  float* bzrx = af(2 * H);
  float* bzrh = af(2 * H);
  float* bzrc = af(2 * H);
  // per-step fp32/bf16 scratch
  float* s_f32   = af((size_t)B * H);
  bf16*  s_bf    = abf((size_t)B * H);
  float* datt    = af((size_t)B * AD);
  float* eraw    = af((size_t)S * B);
  float* invsum  = af(1024);
  float* attendf = af((size_t)B * E2);
  bf16*  attendb = abf((size_t)B * E2);
  float* zr      = af((size_t)B * 2 * H);
  float* hacc    = af((size_t)B * H);
  bf16*  rs_bf   = abf((size_t)B * H);
  float* logit   = af((size_t)B * 2 * OD);

  const dim3 blk(256);
  auto cblocks = [](size_t n) { return (unsigned)((n + 255) / 256); };

  // ---- per-call weight conversion / packing ----
  cvt_f32_bf16<<<cblocks((size_t)AD * H), blk, 0, stream>>>(sa_w, sa_bf, AD * H);
  cvt_f32_bf16<<<cblocks((size_t)H * WD), blk, 0, stream>>>(xz_w, zrx_bf, H * WD);
  cvt_f32_bf16<<<cblocks((size_t)H * WD), blk, 0, stream>>>(xr_w, zrx_bf + (size_t)H * WD, H * WD);
  cvt_f32_bf16<<<cblocks((size_t)H * H), blk, 0, stream>>>(hz_w, zrh_bf, H * H);
  cvt_f32_bf16<<<cblocks((size_t)H * H), blk, 0, stream>>>(hr_w, zrh_bf + (size_t)H * H, H * H);
  cvt_f32_bf16<<<cblocks((size_t)H * E2), blk, 0, stream>>>(cz_w, zrc_bf, H * E2);
  cvt_f32_bf16<<<cblocks((size_t)H * E2), blk, 0, stream>>>(cr_w, zrc_bf + (size_t)H * E2, H * E2);
  cvt_f32_bf16<<<cblocks((size_t)H * WD), blk, 0, stream>>>(xh_w, xh_bf, H * WD);
  cvt_f32_bf16<<<cblocks((size_t)H * H), blk, 0, stream>>>(hh_w, hh_bf, H * H);
  cvt_f32_bf16<<<cblocks((size_t)H * E2), blk, 0, stream>>>(ch_w, ch_bf, H * E2);
  cvt_f32_bf16<<<cblocks((size_t)2 * OD * H), blk, 0, stream>>>(ls_w, ls_bf, 2 * OD * H);
  cvt_f32_bf16<<<cblocks((size_t)2 * OD * WD), blk, 0, stream>>>(ly_w, ly_bf, 2 * OD * WD);
  cvt_f32_bf16<<<cblocks((size_t)2 * OD * E2), blk, 0, stream>>>(lc_w, lc_bf, 2 * OD * E2);
  pack2_f32<<<cblocks(H), blk, 0, stream>>>(xz_b, xr_b, bzrx, H);
  pack2_f32<<<cblocks(H), blk, 0, stream>>>(hz_b, hr_b, bzrh, H);
  pack2_f32<<<cblocks(H), blk, 0, stream>>>(cz_b, cr_b, bzrc, H);
  embed_gather<<<cblocks((size_t)T * B * WD), blk, 0, stream>>>(ys, emb, ye_bf);
  init_state<<<cblocks((size_t)B * H), blk, 0, stream>>>(s_tm1, s_f32, s_bf);

  auto gemm = [&](const bf16* Am, const bf16* Wm, const float* bias,
                  const float* add, float* C, bf16* Cb, int N, int K, int act) {
    dim3 grid(N / 256, B / 16);
    size_t shmem = (size_t)16 * (size_t)K * 2;   // A stripe [16 x K] bf16
    gemm_bf16_wmma<<<grid, blk, shmem, stream>>>(Am, Wm, bias, add, C, Cb,
                                                 B, N, K, act);
  };

  // ---- sequential decode ----
  for (int t = 0; t < T; ++t) {
    const bf16* ye_t = ye_bf + (size_t)t * B * WD;

    // attention
    gemm(s_bf, sa_bf, sa_b, nullptr, datt, nullptr, AD, H, 0);            // [B,A]
    attn_energy<<<S * B / 8, blk, 0, stream>>>(datt, uh, a1_w, a1_b, eraw);
    attn_colsum<<<B / 8, blk, 0, stream>>>(eraw, invsum);
    attn_ctx<<<B * E2 / 256, blk, 0, stream>>>(eraw, invsum, xs_h, attendf, attendb);

    // z,r gates (packed [B,2H])
    gemm(ye_t,    zrx_bf, bzrx, nullptr, zr, nullptr, 2 * H, WD, 0);
    gemm(s_bf,    zrh_bf, bzrh, zr,      zr, nullptr, 2 * H, H,  0);
    gemm(attendb, zrc_bf, bzrc, zr,      zr, nullptr, 2 * H, E2, 1);      // sigmoid
    rs_mul<<<B * H / 256, blk, 0, stream>>>(zr, s_f32, rs_bf);

    // candidate h
    gemm(ye_t,    xh_bf, xh_b, nullptr, hacc, nullptr, H, WD, 0);
    gemm(rs_bf,   hh_bf, hh_b, hacc,    hacc, nullptr, H, H,  0);
    gemm(attendb, ch_bf, ch_b, hacc,    hacc, nullptr, H, E2, 2);         // tanh

    // state update (also writes states output)
    gru_update<<<B * H / 256, blk, 0, stream>>>(zr, hacc, s_f32, s_bf,
                                                d_states + (size_t)t * B * H);

    // maxout readout
    gemm(s_bf,    ls_bf, ls_b, nullptr, logit, nullptr, 2 * OD, H,  0);
    gemm(ye_t,    ly_bf, ly_b, logit,   logit, nullptr, 2 * OD, WD, 0);
    gemm(attendb, lc_bf, lc_b, logit,   logit, nullptr, 2 * OD, E2, 0);
    maxout_mask<<<B * OD / 256, blk, 0, stream>>>(logit, ys_mask + (size_t)t * B,
                                                  d_logits + (size_t)t * B * OD);
  }
}
